// RegionAlignedSNN_69965017252355
// MI455X (gfx1250) — matrix operand order repaired and checked by reading, not compiled, not run on hardware
//
#include <hip/hip_runtime.h>

// ---------------- problem constants ----------------
#define Bsz 32
#define Tlen 200
#define Rn 512
#define Fn 128
#define BRn 4
#define Cn 10
#define Nn 2048          // R*BR
#define Kh 512           // half of 2R (input part / recurrent part)
#define MROWS 6400       // B*T rows of the hoisted input-projection GEMM

typedef __attribute__((ext_vector_type(16))) _Float16 v16h;
typedef __attribute__((ext_vector_type(8)))  _Float16 v8h;
typedef __attribute__((ext_vector_type(8)))  float    v8f;
typedef __attribute__((ext_vector_type(4)))  float    v4f;

// ---------------- workspace layout (bytes) ----------------
constexpr size_t WS_CURPROJ = 0;                                    // 6400*2048 f32 (frag layout)
constexpr size_t WS_CURH    = WS_CURPROJ + (size_t)MROWS * Nn * 4;  // 6400*512 f16, row = t*32+b
constexpr size_t WS_WA      = WS_CURH    + (size_t)MROWS * Kh * 2;  // input-part masked weights f16 [2048][512]
constexpr size_t WS_WB      = WS_WA      + (size_t)Nn * Kh * 2;     // recurrent-part masked weights f16 [2048][512]
constexpr size_t WS_SPK0    = WS_WB      + (size_t)Nn * Kh * 2;     // spike ping buffer f16 [32][512]
constexpr size_t WS_SPK1    = WS_SPK0    + (size_t)Bsz * Rn * 2;    // spike pong buffer
constexpr size_t WS_BAR     = WS_SPK1    + (size_t)Bsz * Rn * 2;    // grid barrier counter
constexpr size_t WS_NEEDED  = WS_BAR + 256;

__device__ __forceinline__ v16h cat8(v8h lo, v8h hi) {
    return __builtin_shufflevector(lo, hi, 0,1,2,3,4,5,6,7,8,9,10,11,12,13,14,15);
}
__device__ __forceinline__ float sigm(float x) { return 1.0f / (1.0f + __expf(-x)); }

// ---------------- phase 0: mask + split + f16-cast recurrent weights ----------------
__global__ __launch_bounds__(256) void prep_w_kernel(const float* __restrict__ w_rnn,
                                                     const float* __restrict__ mask,
                                                     _Float16* __restrict__ Wa,
                                                     _Float16* __restrict__ Wb) {
    size_t i = (size_t)blockIdx.x * 256 + threadIdx.x;   // over 2048*1024
    if (i >= (size_t)Nn * 2 * Kh) return;
    int n = (int)(i >> 10);
    int k = (int)(i & 1023);
    _Float16 v = (_Float16)(w_rnn[i] * mask[i]);
    if (k < Kh) Wa[(size_t)n * Kh + k] = v;
    else        Wb[(size_t)n * Kh + (k - Kh)] = v;
}

// ---------------- phase 1: currents[t*32+b][r] = f16( X[b,t,r,:] . w_in + b_in ) ----------------
__global__ __launch_bounds__(256) void currents_kernel(const float* __restrict__ x,
                                                       const float* __restrict__ w_in,
                                                       const float* __restrict__ b_in,
                                                       _Float16* __restrict__ curh) {
    const int lane = threadIdx.x & 31;
    const int wid  = threadIdx.x >> 5;
    size_t idx = (size_t)blockIdx.x * 8 + wid;           // flat (b,t,r): b*T*R + t*R + r
    const float* row = x + idx * Fn;
    v4f xv = *(const v4f*)(row + lane * 4);
    v4f wv = *(const v4f*)(w_in + lane * 4);
    float s = xv[0]*wv[0] + xv[1]*wv[1] + xv[2]*wv[2] + xv[3]*wv[3];
    s += __shfl_xor(s, 16, 32);
    s += __shfl_xor(s,  8, 32);
    s += __shfl_xor(s,  4, 32);
    s += __shfl_xor(s,  2, 32);
    s += __shfl_xor(s,  1, 32);
    if (lane == 0) {
        int r = (int)(idx % Rn);
        int t = (int)((idx / Rn) % Tlen);
        int b = (int)(idx / ((size_t)Rn * Tlen));
        curh[((size_t)t * Bsz + b) * Kh + r] = (_Float16)(s + b_in[0]);
    }
}

// ---------------- phase 2: curproj = curh[6400,512] @ Wa[2048,512]^T + b_rnn  (WMMA) ----------------
__global__ __launch_bounds__(256) void inproj_kernel(const _Float16* __restrict__ curh,
                                                     const _Float16* __restrict__ Wa,
                                                     const float* __restrict__ b_rnn,
                                                     float* __restrict__ curproj) {
    const int lane = threadIdx.x & 31;
    const int wid  = threadIdx.x >> 5;
    const int tile = blockIdx.x * 8 + wid;     // 0..51199 = 400 mTiles * 128 nTiles
    const int mT   = tile >> 7;
    const int nT   = tile & 127;
    const int nloc = lane & 15;
    const int half = lane >> 4;
    const int ng   = nT * 16 + nloc;

    float bb = b_rnn[ng];
    v8f acc = {bb, bb, bb, bb, bb, bb, bb, bb};

    const _Float16* arow = curh + (size_t)(mT * 16 + nloc) * Kh;
    const _Float16* wrow = Wa   + (size_t)ng * Kh;
#pragma unroll
    for (int kt = 0; kt < 16; ++kt) {
        v8h alo = *(const v8h*)(arow + kt * 32 + half * 8);
        v8h ahi = *(const v8h*)(arow + kt * 32 + 16 + half * 8);
        v8h blo = *(const v8h*)(wrow + kt * 32 + half * 8);
        v8h bhi = *(const v8h*)(wrow + kt * 32 + 16 + half * 8);
        v16h a = cat8(alo, ahi);
        v16h b = cat8(blo, bhi);
        acc = __builtin_amdgcn_wmma_f32_16x16x32_f16(false, a, false, b, (short)0, acc, false, false);
    }
    // store in C-fragment layout: [(mT*128+nT)*32 + lane] * 8 floats, contiguous per lane
    *(v8f*)(curproj + ((size_t)(mT * 128 + nT) * 32 + lane) * 8) = acc;
}

// ---------------- init: zero spike buffers + barrier ----------------
__global__ __launch_bounds__(256) void init_kernel(_Float16* spk0, _Float16* spk1, unsigned* bar) {
    for (int j = threadIdx.x; j < Bsz * Rn; j += 256) { spk0[j] = (_Float16)0.f; spk1[j] = (_Float16)0.f; }
    if (threadIdx.x == 0) bar[0] = 0u;
}

// ---------------- grid barrier (monotonic counter, 32 resident blocks) ----------------
__device__ __forceinline__ void grid_barrier(unsigned* bar, unsigned want) {
    __threadfence();
    __syncthreads();
    if (threadIdx.x == 0) {
        atomicAdd(bar, 1u);
        while (atomicAdd(bar, 0u) < want) { __builtin_amdgcn_s_sleep(2); }
        __threadfence();
    }
    __syncthreads();
}

// ---------------- phase 3: persistent sequential scan, register-resident W_rec ----------------
__global__ __launch_bounds__(256) void scan_kernel(const float* __restrict__ curproj,
                                                   const float* __restrict__ gating,
                                                   const float* __restrict__ tau_m,
                                                   const float* __restrict__ tau_n,
                                                   const float* __restrict__ w_ro,
                                                   const float* __restrict__ b_ro,
                                                   const float* __restrict__ tau_m_ro,
                                                   const _Float16* __restrict__ Wb,
                                                   _Float16* __restrict__ spk0,
                                                   _Float16* __restrict__ spk1,
                                                   unsigned* __restrict__ bar,
                                                   float* __restrict__ out_mem,
                                                   float* __restrict__ out_spk) {
    const int tid   = threadIdx.x;
    const int lane  = tid & 31;
    const int wid   = tid >> 5;
    const int w     = blockIdx.x * 8 + wid;    // 0..255 tiles
    const int nTile = w >> 1;                  // 0..127
    const int mTile = w & 1;                   // batch half
    const int nloc  = lane & 15;
    const int half  = lane >> 4;
    const int ng    = nTile * 16 + nloc;       // neuron-branch column 0..2047
    const int r     = ng >> 2;
    const int br    = ng & 3;

    const float beta  = sigm(tau_n[r * BRn + br]);
    const float alpha = sigm(tau_m[r]);

    // preload B fragments (recurrent weights, K = 512 -> 16 k-tiles), register resident
    v16h breg[16];
    {
        const _Float16* wrow = Wb + (size_t)ng * Kh;
#pragma unroll
        for (int kt = 0; kt < 16; ++kt) {
            v8h lo = *(const v8h*)(wrow + kt * 32 + half * 8);
            v8h hi = *(const v8h*)(wrow + kt * 32 + 16 + half * 8);
            breg[kt] = cat8(lo, hi);
        }
    }

    v8f dstate = {0,0,0,0,0,0,0,0};
    v8f mem    = {0,0,0,0,0,0,0,0};
    v8f spkv   = {0,0,0,0,0,0,0,0};

    // readout state (block 0 only): thread covers up to 2 (b,c) pairs
    float ro_mem[2] = {0.f, 0.f}, ro_spk[2] = {0.f, 0.f};
    float ro_alpha[2] = {0.f, 0.f}, ro_bias[2] = {0.f, 0.f};
    int   ro_b[2] = {0, 0}, ro_c[2] = {0, 0};
    bool  ro_act[2] = {false, false};
    if (blockIdx.x == 0) {
#pragma unroll
        for (int p = 0; p < 2; ++p) {
            int idx = tid + p * 256;
            if (idx < Bsz * Cn) {
                ro_act[p]   = true;
                ro_b[p]     = idx & 31;
                ro_c[p]     = idx >> 5;
                ro_alpha[p] = sigm(tau_m_ro[ro_c[p]]);
                ro_bias[p]  = b_ro[ro_c[p]];
            }
        }
    }

    _Float16* bufs[2] = {spk0, spk1};

    for (int t = 0; t < Tlen; ++t) {
        const _Float16* sprev = bufs[t & 1];
        _Float16*       snext = bufs[(t + 1) & 1];

        // proj = curproj(t) + spikes(t-1) @ Wrec^T   (C-fragment, includes b_rnn)
        v8f acc = *(const v8f*)(curproj + ((size_t)((t * 2 + mTile) * 128 + nTile) * 32 + lane) * 8);
        const _Float16* arow = sprev + (size_t)(mTile * 16 + nloc) * Kh;
#pragma unroll
        for (int kt = 0; kt < 16; ++kt) {
            v8h lo = *(const v8h*)(arow + kt * 32 + half * 8);
            v8h hi = *(const v8h*)(arow + kt * 32 + 16 + half * 8);
            v16h a = cat8(lo, hi);
            acc = __builtin_amdgcn_wmma_f32_16x16x32_f16(false, a, false, breg[kt], (short)0, acc, false, false);
        }

        // dendrite update + branch sum (4 adjacent n-lanes share one neuron r)
        v8f dnew, lin;
#pragma unroll
        for (int v = 0; v < 8; ++v) dnew[v] = beta * dstate[v] + (1.f - beta) * acc[v];
#pragma unroll
        for (int v = 0; v < 8; ++v) {
            float s = dnew[v];
            s += __shfl_xor(s, 1, 32);
            s += __shfl_xor(s, 2, 32);
            lin[v] = s;
        }

        // soma + gating (C layout: row m = mTile*16 + half*8 + v)
#pragma unroll
        for (int v = 0; v < 8; ++v) {
            int m = mTile * 16 + half * 8 + v;
            float mn = mem[v] * alpha + (1.f - alpha) * lin[v] - 0.5f * spkv[v];
            float sn = (mn - 0.5f) > 0.f ? 1.f : 0.f;
            float g  = gating[(size_t)m * (Tlen * Rn) + (size_t)t * Rn + r];
            dstate[v] = g * dnew[v] + (1.f - g) * dstate[v];
            mem[v]    = g * mn      + (1.f - g) * mem[v];
            spkv[v]   = g * sn      + (1.f - g) * spkv[v];
            if (br == 0) {
                snext[(size_t)m * Kh + r] = (_Float16)spkv[v];
                out_spk[(size_t)m * (Rn * Tlen) + (size_t)r * Tlen + t] = spkv[v];
            }
        }

        grid_barrier(bar, 32u * (unsigned)(t + 1));

        // readout LIF on spikes(t) (block 0)
        if (blockIdx.x == 0) {
#pragma unroll
            for (int p = 0; p < 2; ++p) {
                if (ro_act[p]) {
                    const _Float16* sp = snext + (size_t)ro_b[p] * Kh;
                    const float*    wr = w_ro + (size_t)ro_c[p] * Rn;
                    float a2 = 0.f;
                    for (int k = 0; k < Rn; k += 8) {
                        v8h sv = *(const v8h*)(sp + k);
                        v4f w0 = *(const v4f*)(wr + k);
                        v4f w1 = *(const v4f*)(wr + k + 4);
#pragma unroll
                        for (int u = 0; u < 4; ++u) a2 += (float)sv[u]     * w0[u];
#pragma unroll
                        for (int u = 0; u < 4; ++u) a2 += (float)sv[u + 4] * w1[u];
                    }
                    float ro = a2 + ro_bias[p];
                    float ao = ro_alpha[p];
                    float mo = ro_mem[p] * ao + (1.f - ao) * ro - 0.5f * ro_spk[p];
                    ro_spk[p] = (mo - 0.5f) > 0.f ? 1.f : 0.f;
                    ro_mem[p] = mo;
                    out_mem[(size_t)ro_b[p] * (Cn * Tlen) + (size_t)ro_c[p] * Tlen + t] = mo;
                }
            }
        }
    }
}

// ---------------- host launcher ----------------
extern "C" void kernel_launch(void* const* d_in, const int* in_sizes, int n_in,
                              void* d_out, int out_size, void* d_ws, size_t ws_size,
                              hipStream_t stream) {
    (void)in_sizes; (void)n_in; (void)out_size;
    if (ws_size < WS_NEEDED) return;

    const float* x        = (const float*)d_in[0];
    const float* gating   = (const float*)d_in[1];
    const float* w_in     = (const float*)d_in[2];
    const float* b_in     = (const float*)d_in[3];
    const float* w_rnn    = (const float*)d_in[4];
    const float* b_rnn    = (const float*)d_in[5];
    const float* tau_m    = (const float*)d_in[6];
    const float* tau_n    = (const float*)d_in[7];
    const float* w_ro     = (const float*)d_in[8];
    const float* b_ro     = (const float*)d_in[9];
    const float* tau_m_ro = (const float*)d_in[10];
    const float* mask     = (const float*)d_in[11];

    char* ws = (char*)d_ws;
    float*    curproj = (float*)   (ws + WS_CURPROJ);
    _Float16* curh    = (_Float16*)(ws + WS_CURH);
    _Float16* Wa      = (_Float16*)(ws + WS_WA);
    _Float16* Wb      = (_Float16*)(ws + WS_WB);
    _Float16* spk0    = (_Float16*)(ws + WS_SPK0);
    _Float16* spk1    = (_Float16*)(ws + WS_SPK1);
    unsigned* bar     = (unsigned*)(ws + WS_BAR);

    float* out_mem = (float*)d_out;                      // [B, C, T] = 64000 floats
    float* out_spk = out_mem + (size_t)Bsz * Cn * Tlen;  // [B, R, T]

    prep_w_kernel<<<(Nn * 2 * Kh + 255) / 256, 256, 0, stream>>>(w_rnn, mask, Wa, Wb);
    currents_kernel<<<(Bsz * Tlen * Rn) / 8, 256, 0, stream>>>(x, w_in, b_in, curh);
    inproj_kernel<<<(400 * 128) / 8, 256, 0, stream>>>(curh, Wa, b_rnn, curproj);
    init_kernel<<<1, 256, 0, stream>>>(spk0, spk1, bar);
    scan_kernel<<<32, 256, 0, stream>>>(curproj, gating, tau_m, tau_n, w_ro, b_ro, tau_m_ro,
                                        Wb, spk0, spk1, bar, out_mem, out_spk);
}
